// LocalHyperGATlayer_7035156431056
// MI455X (gfx1250) — compile-verified, matrix operand order checked by reading, not run on hardware
//
#include <hip/hip_runtime.h>
#include <hip/hip_bf16.h>

#define BB 256
#define NN 128
#define EE 128
#define DD 128
#define NEGV (-9e15f)
#define SLOPE 0.2f

typedef __attribute__((ext_vector_type(8)))  unsigned short v8s;
typedef __attribute__((ext_vector_type(16))) unsigned short v16s;
typedef __attribute__((ext_vector_type(16))) __bf16         v16bf;
typedef __attribute__((ext_vector_type(8)))  float          v8f;

__device__ __forceinline__ unsigned short f2bf(float f) {
    unsigned u = __builtin_bit_cast(unsigned, f);
    u += 0x7FFFu + ((u >> 16) & 1u);            // round-to-nearest-even
    return (unsigned short)(u >> 16);
}
__device__ __forceinline__ float bf2f(unsigned short s) {
    unsigned u = ((unsigned)s) << 16;
    return __builtin_bit_cast(float, u);
}

// Load one 16x(K=32) bf16 WMMA fragment from a row-major [rows][128] ushort LDS
// array. Per ISA 7.12.2: lane L (row/col = L%16) holds K = klo..klo+7 and
// klo+16..klo+23 where klo = kt*32 + (L>=16 ? 8 : 0). Same layout serves the
// A operand (rows of A) and the B operand (rows of B^T).
__device__ __forceinline__ v16bf ldfrag(const unsigned short* p) {
    v8s lo = *(const v8s*)p;
    v8s hi = *(const v8s*)(p + 16);
    v16s c = __builtin_shufflevector(lo, hi, 0,1,2,3,4,5,6,7,8,9,10,11,12,13,14,15);
    return __builtin_bit_cast(v16bf, c);
}

// C(128x128) row-block m0..m0+15 : acc[t] = A[m0:m0+16,:] * B[:, 16t:16t+16]
// A: row-major [128][128] bf16 ; BT: row-major [128][128] bf16 with BT[n][k]=B[k][n]
__device__ __forceinline__ void gemm_row(const unsigned short* A, const unsigned short* BT,
                                         int lm, int hi, int m0, v8f acc[8]) {
    const v8f vz = {0.f,0.f,0.f,0.f,0.f,0.f,0.f,0.f};
#pragma unroll
    for (int t = 0; t < 8; ++t) acc[t] = vz;
#pragma unroll
    for (int kt = 0; kt < 4; ++kt) {
        const int klo = kt * 32 + hi * 8;
        v16bf a = ldfrag(A + (m0 + lm) * 128 + klo);
#pragma unroll
        for (int t = 0; t < 8; ++t) {
            v16bf bf = ldfrag(BT + (t * 16 + lm) * 128 + klo);
            acc[t] = __builtin_amdgcn_wmma_f32_16x16x32_bf16(
                false, a, false, bf, (short)0, acc[t], false, false);
        }
    }
}

// Row softmax over a 128x128 f32 buffer (in place exp), bf16 result to outbf.
__device__ __forceinline__ void softmax_rows(float* buf, unsigned short* outbf, int tid) {
    if (tid < 128) {
        float* row = buf + tid * 128;
        float m = row[0];
        for (int i = 1; i < 128; ++i) m = fmaxf(m, row[i]);
        float s = 0.f;
        for (int i = 0; i < 128; ++i) { float e = __expf(row[i] - m); row[i] = e; s += e; }
        float inv = 1.0f / s;
        for (int i = 0; i < 128; ++i) outbf[tid * 128 + i] = f2bf(row[i] * inv);
    }
}

// Dynamic LDS carve-up (total 315392 B <= 320 KB/WGP)
#define OFF_HSC    0          // float [128*128]  : h (f32) aliased with score buffer
#define OFF_HADJT  65536      // ushort[128*128]  : H_adj^T bf16 (E,N)
#define OFF_HBF    98304      // ushort[128*128]  : h bf16 (N,D)
#define OFF_HT     131072     // ushort[128*128]  : h^T bf16 (D,N)
#define OFF_BASE   163840     // ushort[128*128]  : ec / edge bf16 (E,D)
#define OFF_SCALED 196608     // ushort[128*128]  : ec*A1k / edge*A2k bf16 (E,D)
#define OFF_EDGET  229376     // ushort[128*128]  : edge^T bf16 (D,E)
#define OFF_ALPHA  262144     // ushort[128*128]  : alpha1^T (E,N) / alpha2 (N,E) bf16
#define OFF_HM     294912     // uchar [128*128]  : edge types (N,E)
#define OFF_SVEC   311296     // float [128]      : s_c row
#define OFF_CORR   311808     // float [128]      : rank-1 (h+s) correction
#define OFF_AS     312320     // float [768]      : A1 (3x128) then A2 (3x128), col-major
#define SMEM_BYTES 315392

__global__ void __launch_bounds__(256)
hypergat_fused(const float* __restrict__ hidden, const float* __restrict__ s_c,
               const float* __restrict__ A1, const float* __restrict__ A2,
               const int* __restrict__ Hg, float* __restrict__ out) {
    extern __shared__ unsigned char smem[];
    float*          hsc    = (float*)(smem + OFF_HSC);
    unsigned short* HadjT  = (unsigned short*)(smem + OFF_HADJT);
    unsigned short* hbf    = (unsigned short*)(smem + OFF_HBF);
    unsigned short* hT     = (unsigned short*)(smem + OFF_HT);
    unsigned short* basebf = (unsigned short*)(smem + OFF_BASE);
    unsigned short* scaled = (unsigned short*)(smem + OFF_SCALED);
    unsigned short* edgeT  = (unsigned short*)(smem + OFF_EDGET);
    unsigned short* alphaA = (unsigned short*)(smem + OFF_ALPHA);
    unsigned char*  Hm     = (unsigned char*)(smem + OFF_HM);
    float*          svec   = (float*)(smem + OFF_SVEC);
    float*          corr   = (float*)(smem + OFF_CORR);
    float*          As     = (float*)(smem + OFF_AS);

    const int tid  = threadIdx.x;
    const int b    = blockIdx.x;
    const int lane = tid & 31;
    const int wv   = tid >> 5;          // 8 wave32 -> 8 tile rows
    const int lm   = lane & 15;
    const int hi   = lane >> 4;

    // ---- per-batch init ----
    for (int i = tid; i < NN * DD; i += 256) hsc[i] = hidden[(size_t)b * NN * DD + i];
    for (int i = tid; i < DD; i += 256)      svec[i] = s_c[(size_t)b * DD + i];
    for (int i = tid; i < 3 * DD; i += 256) {           // As[k][d] = A1[d][k]
        int k = i >> 7, d = i & 127;
        As[k * 128 + d]       = A1[d * 3 + k];
        As[384 + k * 128 + d] = A2[d * 3 + k];
    }
    for (int i = tid; i < NN * EE; i += 256) {
        int v = Hg[(size_t)b * NN * EE + i];
        Hm[i] = (unsigned char)v;
        int n = i >> 7, e = i & 127;
        HadjT[e * NN + n] = (v > 0) ? (unsigned short)0x3F80 : (unsigned short)0;
    }
    __syncthreads();

    for (int layer = 0; layer < 2; ++layer) {
        // P1: h (f32) -> hbf (N,D) and hT (D,N)
        for (int i = tid; i < NN * DD; i += 256) {
            unsigned short u = f2bf(hsc[i]);
            hbf[i] = u;
            int n = i >> 7, d = i & 127;
            hT[d * NN + n] = u;
        }
        __syncthreads();

        // P2: ec(E,D) = HadjT(E,N) @ h(N,D)     [B^T = hT]
        {
            v8f acc[8];
            gemm_row(HadjT, hT, lm, hi, wv * 16, acc);
#pragma unroll
            for (int t = 0; t < 8; ++t)
#pragma unroll
                for (int r = 0; r < 8; ++r) {
                    int e = wv * 16 + r + 8 * hi, d = t * 16 + lm;
                    basebf[e * 128 + d] = f2bf(acc[t][r]);
                }
        }
        __syncthreads();

        // P3: scores1^T (E,N), select-by-type + leaky, masked per k
        for (int i = tid; i < EE * NN; i += 256) hsc[i] = NEGV;
        __syncthreads();
        for (int k = 0; k < 3; ++k) {
            for (int i = tid; i < EE * DD; i += 256)
                scaled[i] = f2bf(bf2f(basebf[i]) * As[k * 128 + (i & 127)]);
            __syncthreads();
            v8f acc[8];
            gemm_row(scaled, hbf, lm, hi, wv * 16, acc);   // scT_k = (ec*A1k) @ h^T
#pragma unroll
            for (int t = 0; t < 8; ++t)
#pragma unroll
                for (int r = 0; r < 8; ++r) {
                    int e = wv * 16 + r + 8 * hi, n = t * 16 + lm;
                    if (Hm[n * 128 + e] == (unsigned char)(k + 1)) {
                        float v = acc[t][r];
                        hsc[e * 128 + n] = v > 0.f ? v : SLOPE * v;
                    }
                }
            __syncthreads();
        }

        // P4: softmax over N (rows of scT) -> alpha1^T bf16 (E,N)
        softmax_rows(hsc, alphaA, tid);
        __syncthreads();

        // P5: edge(E,D) = alpha1^T(E,N) @ h(N,D); also edge^T
        {
            v8f acc[8];
            gemm_row(alphaA, hT, lm, hi, wv * 16, acc);
#pragma unroll
            for (int t = 0; t < 8; ++t)
#pragma unroll
                for (int r = 0; r < 8; ++r) {
                    int e = wv * 16 + r + 8 * hi, d = t * 16 + lm;
                    unsigned short u = f2bf(acc[t][r]);
                    basebf[e * 128 + d] = u;
                    edgeT[d * 128 + e]  = u;
                }
        }
        __syncthreads();

        // P6: scores2 (N,E) with h_t = h + s folded as rank-1 corr[e]
        for (int i = tid; i < NN * EE; i += 256) hsc[i] = NEGV;
        __syncthreads();
        for (int k = 0; k < 3; ++k) {
            for (int i = tid; i < EE * DD; i += 256)
                scaled[i] = f2bf(bf2f(basebf[i]) * As[384 + k * 128 + (i & 127)]);
            if (tid < 128) {                    // corr[e] = sum_d s[d]*A2[d,k]*edge[e,d]
                float c = 0.f;
                for (int d = 0; d < 128; ++d)
                    c += bf2f(basebf[tid * 128 + d]) * As[384 + k * 128 + d] * svec[d];
                corr[tid] = c;
            }
            __syncthreads();
            v8f acc[8];
            gemm_row(hbf, scaled, lm, hi, wv * 16, acc);   // sc2_k = h @ (edge*A2k)^T
#pragma unroll
            for (int t = 0; t < 8; ++t)
#pragma unroll
                for (int r = 0; r < 8; ++r) {
                    int n = wv * 16 + r + 8 * hi, e = t * 16 + lm;
                    if (Hm[n * 128 + e] == (unsigned char)(k + 1)) {
                        float v = acc[t][r] + corr[e];
                        hsc[n * 128 + e] = v > 0.f ? v : SLOPE * v;
                    }
                }
            __syncthreads();
        }

        // P7: softmax over E (rows of sc2) -> alpha2 bf16 (N,E)
        softmax_rows(hsc, alphaA, tid);
        __syncthreads();

        // P8: h_new(N,D) = alpha2(N,E) @ edge(E,D)   [B^T = edgeT]; out += h_new
        {
            v8f acc[8];
            gemm_row(alphaA, edgeT, lm, hi, wv * 16, acc);
#pragma unroll
            for (int t = 0; t < 8; ++t)
#pragma unroll
                for (int r = 0; r < 8; ++r) {
                    int n = wv * 16 + r + 8 * hi, d = t * 16 + lm;
                    float v = acc[t][r];
                    hsc[n * 128 + d] = v;
                    size_t gi = (size_t)b * NN * DD + n * 128 + d;
                    if (layer == 0) out[gi] = v;
                    else            out[gi] += v;
                }
        }
        __syncthreads();
    }
}

extern "C" void kernel_launch(void* const* d_in, const int* in_sizes, int n_in,
                              void* d_out, int out_size, void* d_ws, size_t ws_size,
                              hipStream_t stream) {
    const float* hidden = (const float*)d_in[0];
    const float* s_c    = (const float*)d_in[1];
    const float* A1     = (const float*)d_in[2];
    const float* A2     = (const float*)d_in[3];
    const int*   Hg     = (const int*)d_in[4];
    float* out = (float*)d_out;

    (void)in_sizes; (void)n_in; (void)out_size; (void)d_ws; (void)ws_size;

    // allow >64KB dynamic LDS (MI455X WGP supports 320KB per workgroup)
    hipFuncSetAttribute((const void*)hypergat_fused,
                        hipFuncAttributeMaxDynamicSharedMemorySize, SMEM_BYTES);

    hypergat_fused<<<dim3(BB), dim3(256), SMEM_BYTES, stream>>>(hidden, s_c, A1, A2, Hg, out);
}
